// DCRNNDecoder_34583076668044
// MI455X (gfx1250) — matrix operand order repaired and checked by reading, not compiled, not run on hardware
//
#include <hip/hip_runtime.h>
#include <hip/hip_bf16.h>
#include <math.h>

// ---------------------------------------------------------------------------
// Problem constants (from reference)
// ---------------------------------------------------------------------------
#define NV    325           // graph nodes
#define NVP   336           // padded row stride for supports (multiple of 16)
#define HD    64            // hidden
#define BB    64            // batch
#define MM    5             // 2*K+1 chebyshev mats
#define TT    13            // horizon
#define BN    (BB * NV)     // 20800
#define BNH   (BB * NV * HD)
#define CBMAX (2 * HD * BB) // 8192 (layer-1 C*B)
#define KP0   352           // layer-0 xs K (325) padded to 11*32
#define KP1   640           // layer-1 xs K (20*32, already aligned)

// GEMM tiling
#define BMT 128
#define BNT 128
#define BKT 32
#define AS_STRIDE 40        // padded LDS strides (bf16 elems) to avoid bank conflicts
#define BS_STRIDE 40

typedef __attribute__((ext_vector_type(16))) __bf16 v16bf;
typedef __attribute__((ext_vector_type(8)))  float  v8f;

// ---------------------------------------------------------------------------
// WMMA fragment loaders (wave32 layouts per CDNA5 ISA 7.12.2).
// Both operands laid out in LDS so each lane reads two contiguous 16-byte
// chunks -> ds_load_b128 pairs.
// ---------------------------------------------------------------------------
__device__ inline v16bf load_a_frag(const __bf16* As /*[BMT][AS_STRIDE]*/, int rowOff, int lane) {
    int m  = rowOff + (lane & 15);
    int kb = (lane >> 4) * 8;          // lanes 0-15: K base 0 ; lanes 16-31: K base 8
    const __bf16* p = As + m * AS_STRIDE + kb;
    v16bf a;
#pragma unroll
    for (int j = 0; j < 8; ++j) {
        a[j]     = p[j];               // K = kb..kb+7
        a[j + 8] = p[16 + j];          // K = kb+16..kb+23
    }
    return a;
}

__device__ inline v16bf load_b_frag(const __bf16* Bs /*[BNT][BS_STRIDE] (transposed)*/, int colOff, int lane) {
    int n  = colOff + (lane & 15);
    int kl = (lane >> 4) * 8;          // lanes 0-15: K base 0 ; lanes 16-31: K base 8
    const __bf16* p = Bs + n * BS_STRIDE + kl;
    v16bf b;
#pragma unroll
    for (int j = 0; j < 8; ++j) {
        b[j]     = p[j];               // K = kl..kl+7
        b[j + 8] = p[16 + j];          // K = kl+16..kl+23
    }
    return b;
}

// 16-element (32 B) staged tile loads: vector fast path, scalar ragged edge.
__device__ inline void gload_seg(const __bf16* __restrict__ P, int ldp,
                                 int rows, int cols, int gr, int gc,
                                 __bf16* __restrict__ reg) {
    if (gr < rows && gc + 16 <= cols) {
        const uint4* src = (const uint4*)(P + (size_t)gr * ldp + gc);
        *(uint4*)(reg)     = src[0];
        *(uint4*)(reg + 8) = src[1];
    } else {
#pragma unroll
        for (int j = 0; j < 16; ++j)
            reg[j] = (gr < rows && gc + j < cols) ? P[(size_t)gr * ldp + gc + j]
                                                  : (__bf16)0.0f;
    }
}

// ---------------------------------------------------------------------------
// Generic bf16 GEMM with f32 accumulation:
//   acc = A[MxK] * B[KxN]
//   out = alpha*acc + (bias ? bias[col] : 0) - (sub ? sub[row,col] : 0)
// Block tile 128x128, 8 waves (256 thr); each wave 32x64 via 2x4 WMMA tiles.
// Register-staged double-buffered LDS: next K-tile's global loads are in
// flight while WMMAs execute on the current tile; one barrier per K-step.
// Requires lda % 8 == 0 and ldb % 8 == 0 (callers guarantee via padding).
// ---------------------------------------------------------------------------
__global__ void __launch_bounds__(256)
gemm_bf16_wmma(const __bf16* __restrict__ A, int lda,
               const __bf16* __restrict__ Bm, int ldb,
               int M, int Nn, int K, float alpha,
               const float* __restrict__ sub, int ldsub,
               const float* __restrict__ bias,
               float* __restrict__ outF, int ldof,
               __bf16* __restrict__ outB, int ldob) {
    __shared__ __bf16 As[2][BMT * AS_STRIDE];
    __shared__ __bf16 Bs[2][BNT * BS_STRIDE];

    const int tid  = threadIdx.x;
    const int lane = tid & 31;
    const int wave = tid >> 5;
    const int bm   = blockIdx.y * BMT;
    const int bn   = blockIdx.x * BNT;
    const int wm   = (wave >> 1) * 32;   // 4 row strips of 32
    const int wn   = (wave & 1) * 64;    // 2 col strips of 64

    // fill assignments (16 contiguous elements per thread)
    const int ar  = tid >> 1;            // A: row 0..127
    const int asg = tid & 1;             // A: K segment 0..1
    const int bkr = tid >> 3;            // B: K row 0..31
    const int bsg = tid & 7;             // B: N segment 0..7

    __attribute__((aligned(16))) __bf16 ra[16];
    __attribute__((aligned(16))) __bf16 rb[16];

    // prologue: stage first K tile into registers
    gload_seg(A,  lda, M, K, bm + ar, 0 + asg * 16, ra);
    gload_seg(Bm, ldb, K, Nn, 0 + bkr, bn + bsg * 16, rb);

    v8f acc[2][4] = {};
    int cur = 0;

    for (int k0 = 0; k0 < K; k0 += BKT, cur ^= 1) {
        // commit staged registers to LDS buffer `cur`
        {
            __bf16* dst = &As[cur][ar * AS_STRIDE + asg * 16];
            *(uint4*)(dst)     = *(const uint4*)(ra);
            *(uint4*)(dst + 8) = *(const uint4*)(ra + 8);
        }
#pragma unroll
        for (int j = 0; j < 16; ++j)                       // transpose: Bs[n][k]
            Bs[cur][(bsg * 16 + j) * BS_STRIDE + bkr] = rb[j];
        __syncthreads();

        // kick off next tile's global loads; they overlap the WMMAs below
        if (k0 + BKT < K) {
            gload_seg(A,  lda, M, K,  bm + ar, k0 + BKT + asg * 16, ra);
            gload_seg(Bm, ldb, K, Nn, k0 + BKT + bkr, bn + bsg * 16, rb);
        }

        v16bf af[2], bfrag[4];
#pragma unroll
        for (int i = 0; i < 2; ++i) af[i] = load_a_frag(As[cur], wm + i * 16, lane);
#pragma unroll
        for (int j = 0; j < 4; ++j) bfrag[j] = load_b_frag(Bs[cur], wn + j * 16, lane);

#pragma unroll
        for (int i = 0; i < 2; ++i)
#pragma unroll
            for (int j = 0; j < 4; ++j)
                acc[i][j] = __builtin_amdgcn_wmma_f32_16x16x32_bf16(
                    false, af[i], false, bfrag[j], (short)0, acc[i][j], false, false);
    }

    // epilogue: C/D layout — N' = lane&15 ; M' = v + 8*(lane>>4)
    const int ml = lane & 15;
    const int mh = (lane >> 4) * 8;
#pragma unroll
    for (int i = 0; i < 2; ++i) {
#pragma unroll
        for (int j = 0; j < 4; ++j) {
            int gc = bn + wn + j * 16 + ml;
            if (gc >= Nn) continue;
            float bv = bias ? bias[gc] : 0.0f;
#pragma unroll
            for (int v = 0; v < 8; ++v) {
                int gr = bm + wm + i * 16 + mh + v;
                if (gr >= M) continue;
                float o = alpha * acc[i][j][v] + bv;
                if (sub)  o -= sub[(size_t)gr * ldsub + gc];
                if (outF) outF[(size_t)gr * ldof + gc] = o;
                if (outB) outB[(size_t)gr * ldob + gc] = (__bf16)o;
            }
        }
    }
}

// ---------------------------------------------------------------------------
// Support construction:  S1[i,j] = adj[j,i]/rowsum(j),  S2[i,j] = adj[i,j]/colsum(j)
// stored bf16 with padded row stride NVP (vector-load friendly)
// ---------------------------------------------------------------------------
__global__ void support_sums(const float* __restrict__ adj,
                             float* __restrict__ rowsum, float* __restrict__ colsum) {
    int i = blockIdx.x * blockDim.x + threadIdx.x;
    if (i >= NV) return;
    float rs = 0.f, cs = 0.f;
    for (int j = 0; j < NV; ++j) {
        rs += adj[i * NV + j];
        cs += adj[j * NV + i];
    }
    rowsum[i] = rs;
    colsum[i] = cs;
}

__global__ void support_fill(const float* __restrict__ adj,
                             const float* __restrict__ rowsum, const float* __restrict__ colsum,
                             __bf16* __restrict__ S1b, __bf16* __restrict__ S2b) {
    int idx = blockIdx.x * blockDim.x + threadIdx.x;
    if (idx >= NV * NV) return;
    int i = idx / NV, j = idx % NV;
    S1b[i * NVP + j] = (__bf16)(adj[j * NV + i] / rowsum[j]);
    S2b[i * NVP + j] = (__bf16)(adj[i * NV + j] / colsum[j]);
}

// convert W f32 [rows][cols] -> bf16 [KP][cols], zero pad rows >= rows
__global__ void convert_w_pad(const float* __restrict__ src, int rows, int cols, int KP,
                              __bf16* __restrict__ dst) {
    int idx = blockIdx.x * blockDim.x + threadIdx.x;
    if (idx >= KP * cols) return;
    int r = idx / cols, c = idx % cols;
    dst[idx] = (r < rows) ? (__bf16)src[(size_t)r * cols + c] : (__bf16)0.0f;
}

// ---------------------------------------------------------------------------
// x0 assembly: x0[n, c*B+b] = c < in ? x[b, n*in+c] : h[b, n*HD+(c-in)]
// writes f32 (for Chebyshev sub) and bf16 (GEMM input / mats[0]).
// ---------------------------------------------------------------------------
__global__ void build_x0(const float* __restrict__ x, int in_dim,
                         const float* __restrict__ h, int CB,
                         float* __restrict__ m0f, __bf16* __restrict__ m0b) {
    int idx = blockIdx.x * blockDim.x + threadIdx.x;
    if (idx >= NV * CB) return;
    int n = idx / CB, rem = idx % CB;
    int c = rem / BB, b = rem % BB;
    float v = (c < in_dim) ? x[b * (NV * in_dim) + n * in_dim + c]
                           : h[b * (NV * HD) + n * HD + (c - in_dim)];
    m0f[idx] = v;
    m0b[idx] = (__bf16)v;
}

// xs[b*NV+n, c*MM+m] = mats[m][n, c*B+b]; zero-fill K padding cols
__global__ void pack_xs(const __bf16* __restrict__ mats, int planeStride, int CB,
                        int Kw, int KP, __bf16* __restrict__ xs) {
    int idx = blockIdx.x * blockDim.x + threadIdx.x;
    if (idx >= BN * KP) return;
    int row = idx / KP, col = idx % KP;
    if (col >= Kw) { xs[idx] = (__bf16)0.0f; return; }
    int b = row / NV, n = row % NV;
    int c = col / MM, m = col % MM;
    xs[idx] = mats[(size_t)m * planeStride + n * CB + c * BB + b];
}

// gate activation: r=sig(g[:,0:H]), u=sig(g[:,H:2H]); rh = r*h ; store u
__global__ void gate_act(const float* __restrict__ gate, const float* __restrict__ h,
                         float* __restrict__ rh, float* __restrict__ uu) {
    int idx = blockIdx.x * blockDim.x + threadIdx.x;
    if (idx >= BNH) return;
    int b = idx / (NV * HD), rem = idx % (NV * HD);
    int n = rem / HD, c = rem % HD;
    int row = b * NV + n;
    float r = 1.0f / (1.0f + expf(-gate[(size_t)row * (2 * HD) + c]));
    float u = 1.0f / (1.0f + expf(-gate[(size_t)row * (2 * HD) + HD + c]));
    rh[idx] = r * h[idx];
    uu[idx] = u;
}

// h = u*h + (1-u)*tanh(c_pre)
__global__ void gru_combine(float* __restrict__ h, const float* __restrict__ uu,
                            const float* __restrict__ cpre) {
    int idx = blockIdx.x * blockDim.x + threadIdx.x;
    if (idx >= BNH) return;
    int b = idx / (NV * HD), rem = idx % (NV * HD);
    int n = rem / HD, c = rem % HD;
    int row = b * NV + n;
    float cv = tanhf(cpre[(size_t)row * HD + c]);
    float u  = uu[idx];
    h[idx] = u * h[idx] + (1.0f - u) * cv;
}

// out[b*NV+n] = h[b, n*HD:] . Wfc + bfc
__global__ void fc_out(const float* __restrict__ h, const float* __restrict__ Wfc,
                       const float* __restrict__ bfc, float* __restrict__ out) {
    int idx = blockIdx.x * blockDim.x + threadIdx.x;
    if (idx >= BN) return;
    int b = idx / NV, n = idx % NV;
    const float* hp = h + (size_t)b * (NV * HD) + n * HD;
    float s = bfc[0];
#pragma unroll 8
    for (int c = 0; c < HD; ++c) s += hp[c] * Wfc[c];
    out[idx] = s;
}

__global__ void zero_f32(float* __restrict__ p, int n) {
    int i = blockIdx.x * blockDim.x + threadIdx.x;
    if (i < n) p[i] = 0.0f;
}

// ---------------------------------------------------------------------------
// Host orchestration
// ---------------------------------------------------------------------------
static inline void launch_gemm(const __bf16* A, int lda, const __bf16* B, int ldb,
                               int M, int Nc, int K, float alpha,
                               const float* sub, int ldsub, const float* bias,
                               float* outF, int ldof, __bf16* outB, int ldob,
                               hipStream_t s) {
    dim3 grid((Nc + BNT - 1) / BNT, (M + BMT - 1) / BMT), block(256);
    gemm_bf16_wmma<<<grid, block, 0, s>>>(A, lda, B, ldb, M, Nc, K, alpha,
                                          sub, ldsub, bias, outF, ldof, outB, ldob);
}

extern "C" void kernel_launch(void* const* d_in, const int* in_sizes, int n_in,
                              void* d_out, int out_size, void* d_ws, size_t ws_size,
                              hipStream_t stream) {
    (void)in_sizes; (void)n_in; (void)out_size; (void)ws_size;

    const float* inputs     = (const float*)d_in[0];  // [13,64,325,1]
    const float* init_state = (const float*)d_in[1];  // [2, 64, 325*64]
    const float* adj        = (const float*)d_in[2];  // [325,325]
    const float* Wg0 = (const float*)d_in[3];  const float* bg0 = (const float*)d_in[4];
    const float* Wc0 = (const float*)d_in[5];  const float* bc0 = (const float*)d_in[6];
    const float* Wg1 = (const float*)d_in[7];  const float* bg1 = (const float*)d_in[8];
    const float* Wc1 = (const float*)d_in[9];  const float* bc1 = (const float*)d_in[10];
    const float* Wfc = (const float*)d_in[11]; const float* bfc = (const float*)d_in[12];
    float* out = (float*)d_out;

    // ---- workspace carve-up ----
    char* p = (char*)d_ws;
    auto alloc = [&](size_t bytes) -> void* {
        void* r = (void*)p;
        p += (bytes + 255) & ~(size_t)255;
        return r;
    };
    float*  rowsum = (float*)alloc(NV * sizeof(float));
    float*  colsum = (float*)alloc(NV * sizeof(float));
    __bf16* S1b    = (__bf16*)alloc((size_t)NVP * NVP * 2);
    __bf16* S2b    = (__bf16*)alloc((size_t)NVP * NVP * 2);
    __bf16* Wg0b   = (__bf16*)alloc((size_t)KP0 * 128 * 2);
    __bf16* Wc0b   = (__bf16*)alloc((size_t)KP0 * 64 * 2);
    __bf16* Wg1b   = (__bf16*)alloc((size_t)KP1 * 128 * 2);
    __bf16* Wc1b   = (__bf16*)alloc((size_t)KP1 * 64 * 2);
    float*  hst    = (float*)alloc((size_t)2 * BNH * sizeof(float));       // layer states
    float*  m0f    = (float*)alloc((size_t)NV * CBMAX * sizeof(float));    // x0 (f32) for 2*y - x0
    __bf16* matsB  = (__bf16*)alloc((size_t)MM * NV * CBMAX * 2);          // 5 chebyshev mats (bf16)
    __bf16* xsb    = (__bf16*)alloc((size_t)BN * KP1 * 2);                 // packed xs
    float*  gout   = (float*)alloc((size_t)BN * 2 * HD * sizeof(float));   // gate out / reused as c_pre
    float*  rh     = (float*)alloc((size_t)BNH * sizeof(float));
    float*  uu     = (float*)alloc((size_t)BNH * sizeof(float));
    const size_t PLANE = (size_t)NV * CBMAX;

    // ---- one-time prep (inside launch: deterministic, capture-safe) ----
    hipMemcpyAsync(hst, init_state, (size_t)2 * BNH * sizeof(float),
                   hipMemcpyDeviceToDevice, stream);

    support_sums<<<(NV + 255) / 256, 256, 0, stream>>>(adj, rowsum, colsum);
    support_fill<<<(NV * NV + 255) / 256, 256, 0, stream>>>(adj, rowsum, colsum, S1b, S2b);
    convert_w_pad<<<(KP0 * 128 + 255) / 256, 256, 0, stream>>>(Wg0, 325, 128, KP0, Wg0b);
    convert_w_pad<<<(KP0 * 64  + 255) / 256, 256, 0, stream>>>(Wc0, 325, 64,  KP0, Wc0b);
    convert_w_pad<<<(KP1 * 128 + 255) / 256, 256, 0, stream>>>(Wg1, 640, 128, KP1, Wg1b);
    convert_w_pad<<<(KP1 * 64  + 255) / 256, 256, 0, stream>>>(Wc1, 640, 64,  KP1, Wc1b);
    zero_f32<<<(BN + 255) / 256, 256, 0, stream>>>(out, BN);   // outputs[0] == 0

    // one graph-convolution: mats -> xs -> xs @ W + b  (result f32 [BN, outW])
    auto gconv = [&](const float* x, int in_dim, const float* h,
                     const __bf16* Wb, const float* bias, int outW, int KP, float* res) {
        const int C  = in_dim + HD;
        const int CB = C * BB;
        const int Kw = C * MM;
        // x0 (mats[0])
        build_x0<<<((size_t)NV * CB + 255) / 256, 256, 0, stream>>>(
            x, in_dim, h, CB, m0f, matsB + 0 * PLANE);
        // m1 = S1 @ x0 ; m2 = 2*S1@m1 - x0 ; m3 = S2 @ x0 ; m4 = 2*S2@m3 - x0
        launch_gemm(S1b, NVP, matsB + 0 * PLANE, CB, NV, CB, NV, 1.0f,
                    nullptr, 0, nullptr, nullptr, 0, matsB + 1 * PLANE, CB, stream);
        launch_gemm(S1b, NVP, matsB + 1 * PLANE, CB, NV, CB, NV, 2.0f,
                    m0f, CB, nullptr, nullptr, 0, matsB + 2 * PLANE, CB, stream);
        launch_gemm(S2b, NVP, matsB + 0 * PLANE, CB, NV, CB, NV, 1.0f,
                    nullptr, 0, nullptr, nullptr, 0, matsB + 3 * PLANE, CB, stream);
        launch_gemm(S2b, NVP, matsB + 3 * PLANE, CB, NV, CB, NV, 2.0f,
                    m0f, CB, nullptr, nullptr, 0, matsB + 4 * PLANE, CB, stream);
        // pack xs (zero-padded K) and project with fused bias
        pack_xs<<<((size_t)BN * KP + 255) / 256, 256, 0, stream>>>(
            matsB, (int)PLANE, CB, Kw, KP, xsb);
        launch_gemm(xsb, KP, Wb, outW, BN, outW, KP, 1.0f,
                    nullptr, 0, bias, res, outW, nullptr, 0, stream);
    };

    // ---- time loop (teacher forcing: step t uses inputs[t-1]) ----
    for (int t = 1; t < TT; ++t) {
        const float* x_t = inputs + (size_t)(t - 1) * BN;   // [B, N, 1]

        for (int l = 0; l < 2; ++l) {
            const int    in_dim = (l == 0) ? 1 : HD;
            const float* x_in   = (l == 0) ? x_t : (hst + 0 * BNH);  // layer 1 input = layer 0 new h
            float*       h      = hst + (size_t)l * BNH;
            const __bf16* Wgb   = (l == 0) ? Wg0b : Wg1b;
            const float*  bg    = (l == 0) ? bg0  : bg1;
            const __bf16* Wcb   = (l == 0) ? Wc0b : Wc1b;
            const float*  bc    = (l == 0) ? bc0  : bc1;
            const int     KP    = (l == 0) ? KP0  : KP1;

            // gate gconv -> r, u
            gconv(x_in, in_dim, h, Wgb, bg, 2 * HD, KP, gout);
            gate_act<<<(BNH + 255) / 256, 256, 0, stream>>>(gout, h, rh, uu);
            // candidate gconv (x, r*h) -> c ; h = u*h + (1-u)*tanh(c)
            gconv(x_in, in_dim, rh, Wcb, bc, HD, KP, gout);  // gout reused as c_pre
            gru_combine<<<(BNH + 255) / 256, 256, 0, stream>>>(h, uu, gout);
        }
        // projection of top-layer state
        fc_out<<<(BN + 255) / 256, 256, 0, stream>>>(hst + 1 * BNH, Wfc, bfc,
                                                     out + (size_t)t * BN);
    }
}
